// MultiHeadedEncoderDecoderRelativeAttentionBias_63324997812553
// MI455X (gfx1250) — compile-verified, matrix operand order checked by reading, not run on hardware
//
#include <hip/hip_runtime.h>
#include <hip/hip_bf16.h>

// ---------------------------------------------------------------------------
// MI455X (gfx1250) T5-style cross attention, bf16 WMMA pipeline.
// v_wmma_f32_16x16x32_bf16 for all matmuls, TDM tensor_load_to_lds for tiles,
// b128 LDS fragment loads (contiguous 8-elem runs of the WMMA 16-bit layout).
// ---------------------------------------------------------------------------

typedef __attribute__((ext_vector_type(16))) __bf16 bf16x16;
typedef __attribute__((ext_vector_type(8)))  __bf16 bf16x8;
typedef __attribute__((ext_vector_type(8)))  float  f32x8;
typedef __attribute__((ext_vector_type(4)))  unsigned int u32x4;
typedef __attribute__((ext_vector_type(8)))  int    i32x8;
typedef __attribute__((ext_vector_type(4)))  int    i32x4;

union AccF   { f32x8   v; float  e[8];  };
union Frag16 { bf16x16 v; bf16x8 h[2]; __bf16 e[16]; };

#define D_MODEL 1024
#define HEADS   16
#define D_K     64

// Contiguous-run fragment load: 16-bit A(16x32)/B(32x16) layout means lane
// (hlf = lane>>4) holds K = kb+hlf*8..+7 then K = kb+16+hlf*8..+7  -> 2x b128.
__device__ __forceinline__ bf16x16 frag_ld(const __bf16* rowp, int hlf) {
  Frag16 f;
  f.h[0] = *(const bf16x8*)(rowp + hlf * 8);
  f.h[1] = *(const bf16x8*)(rowp + 16 + hlf * 8);
  return f.v;
}

__device__ __forceinline__ unsigned lds_off(const void* p) {
  return (unsigned)(unsigned long long)p;  // LDS addr = low 32 bits of flat
}

// TDM: 2D tile (tile_d0 contiguous elems x tile_d1 rows) of bf16, row stride
// stride0 elems, from global gptr into LDS offset ldsoff. Issue from one wave.
__device__ __forceinline__ void tdm_load_2d(unsigned ldsoff, const void* gptr,
                                            int tile_d0, int tile_d1,
                                            int tensor_d0, int tensor_d1,
                                            int stride0) {
  unsigned long long ga = (unsigned long long)gptr;
  u32x4 g0;
  g0[0] = 1u;                                            // count=1, user mode
  g0[1] = ldsoff;                                        // lds_addr
  g0[2] = (unsigned)(ga & 0xFFFFFFFFu);                  // global_addr[31:0]
  g0[3] = (unsigned)((ga >> 32) & 0x01FFFFFFu) | (2u << 30);  // [56:32]|type=2
  i32x8 g1;
  g1[0] = (int)(1u << 16);                               // data_size=1 (2B)
  g1[1] = (int)(((unsigned)tensor_d0 & 0xFFFFu) << 16);  // tensor_dim0 lo
  g1[2] = (int)((((unsigned)tensor_d0 >> 16) & 0xFFFFu) |
                (((unsigned)tensor_d1 & 0xFFFFu) << 16));
  g1[3] = (int)((((unsigned)tensor_d1 >> 16) & 0xFFFFu) |
                ((unsigned)tile_d0 << 16));              // tile_dim0
  g1[4] = (int)((unsigned)tile_d1 & 0xFFFFu);            // tile_dim1
  g1[5] = stride0;                                       // tensor_dim0_stride
  g1[6] = 0;
  g1[7] = 0;
  i32x4 z4 = {0, 0, 0, 0};
  i32x8 z8 = {0, 0, 0, 0, 0, 0, 0, 0};
  __builtin_amdgcn_tensor_load_to_lds(g0, g1, z4, z4, z8, 0);
}

// ---------------------------------------------------------------------------
// fp32 -> bf16 downconvert (4 elems/thread, b128 in / b64 out)
// ---------------------------------------------------------------------------
__global__ __launch_bounds__(256) void cast_f32_to_bf16(
    const float* __restrict__ in, __bf16* __restrict__ out, int n4) {
  int i = blockIdx.x * blockDim.x + threadIdx.x;
  if (i < n4) {
    float4 f = ((const float4*)in)[i];
    union { __bf16 b[4]; uint2 u; } o;
    o.b[0] = (__bf16)f.x; o.b[1] = (__bf16)f.y;
    o.b[2] = (__bf16)f.z; o.b[3] = (__bf16)f.w;
    *(uint2*)(&out[i * 4]) = o.u;
  }
}

// ---------------------------------------------------------------------------
// C[M,N] = A[M,K] @ W[N,K]^T   (bf16 in, f32 accum, bf16 or f32 out)
// Block tile 32(M) x 64(N), K-step 64 (2 WMMAs per barrier period).
// ---------------------------------------------------------------------------
template <bool STORE_BF16>
__global__ __launch_bounds__(256) void gemm_bf16_wmma(
    const __bf16* __restrict__ A, const __bf16* __restrict__ W,
    void* __restrict__ Cout, int M, int N, int K) {
  __shared__ __align__(16) __bf16 As[32 * 64];  // [m][k]
  __shared__ __align__(16) __bf16 Ws[64 * 64];  // [n][k]

  const int tid  = threadIdx.x;
  const int lane = tid & 31;
  const int wave = tid >> 5;
  const int hlf  = lane >> 4;
  const int lm   = lane & 15;
  const int msub = wave >> 2;   // 0..1
  const int nsub = wave & 3;    // 0..3
  const int m0   = blockIdx.y * 32;
  const int n0   = blockIdx.x * 64;

  AccF acc;
#pragma unroll
  for (int r = 0; r < 8; ++r) acc.e[r] = 0.0f;

  for (int kk = 0; kk < K; kk += 64) {
    {  // A tile: 32x64 bf16, 8 elems/thread (b128)
      int row = tid >> 3, col = (tid & 7) * 8;
      *(uint4*)(&As[row * 64 + col]) =
          *(const uint4*)(&A[(size_t)(m0 + row) * K + kk + col]);
      if (kk + 64 < K)
        __builtin_prefetch(&A[(size_t)(m0 + row) * K + kk + 64 + col], 0, 0);
    }
    {  // W tile: 64x64 bf16, 16 elems/thread (2x b128)
      int row = tid >> 2, col = (tid & 3) * 16;
      const __bf16* g = &W[(size_t)(n0 + row) * K + kk + col];
      *(uint4*)(&Ws[row * 64 + col])     = *(const uint4*)(g);
      *(uint4*)(&Ws[row * 64 + col + 8]) = *(const uint4*)(g + 8);
      if (kk + 64 < K) __builtin_prefetch(g + 64, 0, 0);
    }
    __syncthreads();

#pragma unroll
    for (int ks = 0; ks < 64; ks += 32) {
      bf16x16 a = frag_ld(&As[(msub * 16 + lm) * 64 + ks], hlf);
      bf16x16 b = frag_ld(&Ws[(nsub * 16 + lm) * 64 + ks], hlf);
      acc.v = __builtin_amdgcn_wmma_f32_16x16x32_bf16(
          false, a, false, b, (short)0, acc.v, false, false);
    }
    __syncthreads();
  }

#pragma unroll
  for (int r = 0; r < 8; ++r) {
    int mg = m0 + msub * 16 + r + 8 * hlf;
    int ng = n0 + nsub * 16 + lm;
    if (STORE_BF16)
      ((__bf16*)Cout)[(size_t)mg * N + ng] = (__bf16)acc.e[r];
    else
      ((float*)Cout)[(size_t)mg * N + ng] = acc.e[r];
  }
}

// ---------------------------------------------------------------------------
// T5 relative position bucket bias (NUM_BUCKETS=32, MAX_DISTANCE=128)
// ---------------------------------------------------------------------------
__device__ __forceinline__ float rel_bias_val(const float* __restrict__ rb,
                                              int h, int rp) {
  const int nb = 16, max_exact = 8;
  int bucket = (rp > 0) ? nb : 0;
  int arp = rp < 0 ? -rp : rp;
  if (arp < max_exact) {
    bucket += arp;
  } else {
    int lg = max_exact + (int)(__logf((float)arp * 0.125f) * 2.885390082f);
    bucket += (lg < nb - 1) ? lg : (nb - 1);
  }
  return rb[h * 32 + bucket];
}

// ---------------------------------------------------------------------------
// Flash attention: block = (b, h, 32 q-rows); 64-key tiles streamed via TDM.
// ---------------------------------------------------------------------------
__global__ __launch_bounds__(256) void attn_flash(
    const __bf16* __restrict__ qb, const __bf16* __restrict__ kb,
    const __bf16* __restrict__ vb, const unsigned char* __restrict__ mask,
    const float* __restrict__ rel_bias, __bf16* __restrict__ ctx,
    int B, int Tq, int Tk) {
  __shared__ __align__(16) __bf16 Qs[32 * 64];   // [q][d]
  __shared__ __align__(16) __bf16 Ks[64 * 64];   // [key][d]
  __shared__ __align__(16) __bf16 Vt[64 * 64];   // [d][key]  (transposed V)
  __shared__ __align__(16) float  Ss[32 * 64];   // [q][key]
  __shared__ float rowmax[32], rowsum[32], rowscale[32];

  const int tid  = threadIdx.x;
  const int lane = tid & 31;
  const int wave = tid >> 5;
  const int hlf  = lane >> 4;
  const int lm   = lane & 15;
  const int msub = wave >> 2;   // 0..1  (16 q-rows each)
  const int nsub = wave & 3;    // 0..3  (16 keys for S / 16 dims for O)

  const int q0 = blockIdx.x * 32;
  const int h  = blockIdx.y;
  const int b  = blockIdx.z;

  // Q tile 32x64 via TDM from wave 0 (tile: 64 contiguous dims x 32 rows)
  if (wave == 0) {
    tdm_load_2d(lds_off(Qs),
                qb + (size_t)(b * Tq + q0) * D_MODEL + h * D_K,
                /*tile_d0=*/D_K, /*tile_d1=*/32,
                /*tensor_d0=*/D_MODEL, /*tensor_d1=*/1 << 20,
                /*stride0=*/D_MODEL);
    __builtin_amdgcn_s_wait_tensorcnt(0);
  }
  if (tid < 32) { rowmax[tid] = -INFINITY; rowsum[tid] = 0.0f; }

  AccF o;
#pragma unroll
  for (int r = 0; r < 8; ++r) o.e[r] = 0.0f;
  __syncthreads();

  for (int kt = 0; kt < Tk; kt += 64) {
    // K tile 64x64 via TDM (async; overlaps with V transpose below)
    if (wave == 0) {
      tdm_load_2d(lds_off(Ks),
                  kb + (size_t)(b * Tk + kt) * D_MODEL + h * D_K,
                  /*tile_d0=*/D_K, /*tile_d1=*/64,
                  /*tensor_d0=*/D_MODEL, /*tensor_d1=*/1 << 20,
                  /*stride0=*/D_MODEL);
    }
    {  // V tile 64x64: b128 global loads + transpose into Vt[d][key]
      int row = tid >> 2, col = (tid & 3) * 16;
      const __bf16* g = &vb[(size_t)(b * Tk + kt + row) * D_MODEL + h * D_K + col];
      union { uint4 u; __bf16 e[8]; } r0, r1;
      r0.u = *(const uint4*)(g);
      r1.u = *(const uint4*)(g + 8);
#pragma unroll
      for (int j = 0; j < 8; ++j) {
        Vt[(col + j) * 64 + row]     = r0.e[j];
        Vt[(col + 8 + j) * 64 + row] = r1.e[j];
      }
    }
    if (wave == 0) __builtin_amdgcn_s_wait_tensorcnt(0);
    __syncthreads();

    // ---- S = Q K^T : wave -> 16(q) x 16(key) subtile, 2 WMMAs over d
    AccF sacc;
#pragma unroll
    for (int r = 0; r < 8; ++r) sacc.e[r] = 0.0f;
#pragma unroll
    for (int ks = 0; ks < 64; ks += 32) {
      bf16x16 a = frag_ld(&Qs[(msub * 16 + lm) * 64 + ks], hlf);
      bf16x16 bb = frag_ld(&Ks[(nsub * 16 + lm) * 64 + ks], hlf);
      sacc.v = __builtin_amdgcn_wmma_f32_16x16x32_bf16(
          false, a, false, bb, (short)0, sacc.v, false, false);
    }
    // bias + mask at C-layout positions, spill to LDS for softmax
#pragma unroll
    for (int r = 0; r < 8; ++r) {
      int qm = msub * 16 + r + 8 * hlf;
      int kn = nsub * 16 + lm;
      int k_idx = kt + kn;
      float val = sacc.e[r] + rel_bias_val(rel_bias, h, k_idx - (q0 + qm));
      if (!mask[b * Tk + k_idx]) val = -1.0e9f;
      Ss[qm * 64 + kn] = val;
    }
    __syncthreads();

    // ---- online softmax stats: 8 lanes per q-row, shfl_xor reductions
    {
      int row = tid >> 3, seg = (tid & 7) * 8;
      const float4* p = (const float4*)&Ss[row * 64 + seg];
      float4 f0 = p[0], f1 = p[1];
      float m8 = fmaxf(fmaxf(fmaxf(f0.x, f0.y), fmaxf(f0.z, f0.w)),
                       fmaxf(fmaxf(f1.x, f1.y), fmaxf(f1.z, f1.w)));
      m8 = fmaxf(m8, __shfl_xor(m8, 1, 8));
      m8 = fmaxf(m8, __shfl_xor(m8, 2, 8));
      m8 = fmaxf(m8, __shfl_xor(m8, 4, 8));
      float mold = rowmax[row];
      float mnew = fmaxf(mold, m8);
      float s8 = __expf(f0.x - mnew) + __expf(f0.y - mnew) +
                 __expf(f0.z - mnew) + __expf(f0.w - mnew) +
                 __expf(f1.x - mnew) + __expf(f1.y - mnew) +
                 __expf(f1.z - mnew) + __expf(f1.w - mnew);
      s8 += __shfl_xor(s8, 1, 8);
      s8 += __shfl_xor(s8, 2, 8);
      s8 += __shfl_xor(s8, 4, 8);
      if ((tid & 7) == 0) {
        float scl = __expf(mold - mnew);
        rowsum[row]   = rowsum[row] * scl + s8;
        rowmax[row]   = mnew;
        rowscale[row] = scl;
      }
    }
    __syncthreads();

    // ---- O = O*scale + P V : wave -> 16(q) x 16(d), 2 WMMAs over keys
#pragma unroll
    for (int r = 0; r < 8; ++r) o.e[r] *= rowscale[msub * 16 + r + 8 * hlf];
    {
      float mrow = rowmax[msub * 16 + lm];
#pragma unroll
      for (int ks = 0; ks < 64; ks += 32) {
        // P fragment: contiguous f32 runs (2x2 b128) + exp, packed to bf16
        Frag16 a;
        const float4* p0 = (const float4*)&Ss[(msub * 16 + lm) * 64 + ks + hlf * 8];
        const float4* p1 = (const float4*)&Ss[(msub * 16 + lm) * 64 + ks + 16 + hlf * 8];
        float4 x0 = p0[0], x1 = p0[1], y0 = p1[0], y1 = p1[1];
        a.e[0]  = (__bf16)__expf(x0.x - mrow); a.e[1]  = (__bf16)__expf(x0.y - mrow);
        a.e[2]  = (__bf16)__expf(x0.z - mrow); a.e[3]  = (__bf16)__expf(x0.w - mrow);
        a.e[4]  = (__bf16)__expf(x1.x - mrow); a.e[5]  = (__bf16)__expf(x1.y - mrow);
        a.e[6]  = (__bf16)__expf(x1.z - mrow); a.e[7]  = (__bf16)__expf(x1.w - mrow);
        a.e[8]  = (__bf16)__expf(y0.x - mrow); a.e[9]  = (__bf16)__expf(y0.y - mrow);
        a.e[10] = (__bf16)__expf(y0.z - mrow); a.e[11] = (__bf16)__expf(y0.w - mrow);
        a.e[12] = (__bf16)__expf(y1.x - mrow); a.e[13] = (__bf16)__expf(y1.y - mrow);
        a.e[14] = (__bf16)__expf(y1.z - mrow); a.e[15] = (__bf16)__expf(y1.w - mrow);
        // V fragment: Vt[d][key] rows -> contiguous in key (2x b128)
        bf16x16 bb = frag_ld(&Vt[(nsub * 16 + lm) * 64 + ks], hlf);
        o.v = __builtin_amdgcn_wmma_f32_16x16x32_bf16(
            false, a.v, false, bb, (short)0, o.v, false, false);
      }
    }
    __syncthreads();
  }

#pragma unroll
  for (int r = 0; r < 8; ++r) {
    int qm = msub * 16 + r + 8 * hlf;
    float val = o.e[r] / rowsum[qm];
    ctx[(size_t)(b * Tq + q0 + qm) * D_MODEL + h * D_K + nsub * 16 + lm] =
        (__bf16)val;
  }
}

// ---------------------------------------------------------------------------
// Host: cast -> Q/K/V GEMMs -> flash attention -> output GEMM
// ---------------------------------------------------------------------------
extern "C" void kernel_launch(void* const* d_in, const int* in_sizes, int n_in,
                              void* d_out, int out_size, void* d_ws, size_t ws_size,
                              hipStream_t stream) {
  (void)in_sizes; (void)n_in; (void)out_size; (void)ws_size;
  const float* src = (const float*)d_in[0];
  const float* dst = (const float*)d_in[1];
  const unsigned char* mask = (const unsigned char*)d_in[2];
  const float* Wq = (const float*)d_in[3];
  const float* Wk = (const float*)d_in[4];
  const float* Wv = (const float*)d_in[5];
  const float* Wo = (const float*)d_in[6];
  const float* rel_bias = (const float*)d_in[7];

  const int B = 4, Tq = 1024, Tk = 2048;
  const size_t nDst = (size_t)B * Tq * D_MODEL;   // 4,194,304
  const size_t nSrc = (size_t)B * Tk * D_MODEL;   // 8,388,608
  const size_t nW   = (size_t)D_MODEL * D_MODEL;  // 1,048,576

  char* ws = (char*)d_ws;
  size_t off = 0;
  auto alloc = [&](size_t bytes) -> void* {
    void* p = ws + off; off += (bytes + 255) & ~(size_t)255; return p;
  };
  __bf16* dst_bf = (__bf16*)alloc(nDst * 2);
  __bf16* src_bf = (__bf16*)alloc(nSrc * 2);
  __bf16* Wq_bf  = (__bf16*)alloc(nW * 2);
  __bf16* Wk_bf  = (__bf16*)alloc(nW * 2);
  __bf16* Wv_bf  = (__bf16*)alloc(nW * 2);
  __bf16* Wo_bf  = (__bf16*)alloc(nW * 2);
  __bf16* q_bf   = (__bf16*)alloc(nDst * 2);
  __bf16* k_bf   = (__bf16*)alloc(nSrc * 2);
  __bf16* v_bf   = (__bf16*)alloc(nSrc * 2);
  __bf16* ctx_bf = (__bf16*)alloc(nDst * 2);

  const int TB = 256;
  auto cast = [&](const float* in, __bf16* out, size_t n) {
    int n4 = (int)(n / 4);
    cast_f32_to_bf16<<<(n4 + TB - 1) / TB, TB, 0, stream>>>(in, out, n4);
  };
  cast(dst, dst_bf, nDst);
  cast(src, src_bf, nSrc);
  cast(Wq, Wq_bf, nW);
  cast(Wk, Wk_bf, nW);
  cast(Wv, Wv_bf, nW);
  cast(Wo, Wo_bf, nW);

  // Projections: q = dst@Wq^T ; k = src@Wk^T ; v = src@Wv^T
  gemm_bf16_wmma<true><<<dim3(D_MODEL / 64, (B * Tq) / 32), TB, 0, stream>>>(
      dst_bf, Wq_bf, q_bf, B * Tq, D_MODEL, D_MODEL);
  gemm_bf16_wmma<true><<<dim3(D_MODEL / 64, (B * Tk) / 32), TB, 0, stream>>>(
      src_bf, Wk_bf, k_bf, B * Tk, D_MODEL, D_MODEL);
  gemm_bf16_wmma<true><<<dim3(D_MODEL / 64, (B * Tk) / 32), TB, 0, stream>>>(
      src_bf, Wv_bf, v_bf, B * Tk, D_MODEL, D_MODEL);

  // Flash attention with T5 relative bias + mask
  attn_flash<<<dim3(Tq / 32, HEADS, B), TB, 0, stream>>>(
      q_bf, k_bf, v_bf, mask, rel_bias, ctx_bf, B, Tq, Tk);

  // Output projection: out = ctx @ Wo^T (fp32 result)
  gemm_bf16_wmma<false><<<dim3(D_MODEL / 64, (B * Tq) / 32), TB, 0, stream>>>(
      ctx_bf, Wo_bf, d_out, B * Tq, D_MODEL, D_MODEL);
}
